// Explicit_corr_618475290901
// MI455X (gfx1250) — compile-verified, matrix operand order checked by reading, not run on hardware
//
#include <hip/hip_runtime.h>
#include <hip/hip_bf16.h>

typedef __attribute__((ext_vector_type(16))) __bf16 v16bf;
typedef __attribute__((ext_vector_type(8)))  float  v8f;
typedef __attribute__((ext_vector_type(4)))  unsigned int u32x4;
typedef __attribute__((ext_vector_type(8)))  int i32x8;
typedef __attribute__((ext_vector_type(4)))  int i32x4;

#define CPITCH 272     // bf16 elements per pixel row in LDS (32B-aligned rows, bank-spread)
#define NPIX   225
#define NPAD   240
#define XS_BYTES   (NPAD*CPITCH*2)          // 130560
#define WS_LDS_OFF XS_BYTES                 // LDS byte offset of staged weights
#define WS_ELEMS   43520                    // W1(32768)+W2(8192)+W3(2048)+W4(512) bf16
#define WS_BYTES   (WS_ELEMS*2)             // 87040
#define MISC_OFF   (XS_BYTES + WS_BYTES)

#if __has_include(<hip/amd_detail/amd_gfx1250_TDM.h>)
#define TDM_6ARG 1
#else
#define TDM_6ARG 0
#endif

__device__ __forceinline__ float leaky(float x){ return x >= 0.f ? x : 0.01f * x; }

struct H8 { unsigned int u[4]; };   // 8 bf16 values packed

__device__ __forceinline__ unsigned int pk2(float a, float b){
  union { __bf16 h[2]; unsigned int u; } t;
  t.h[0] = (__bf16)a; t.h[1] = (__bf16)b;
  return t.u;
}

__device__ __forceinline__ H8 shfl16(H8 s){
  H8 r;
  r.u[0] = __shfl_xor(s.u[0], 16, 32);
  r.u[1] = __shfl_xor(s.u[1], 16, 32);
  r.u[2] = __shfl_xor(s.u[2], 16, 32);
  r.u[3] = __shfl_xor(s.u[3], 16, 32);
  return r;
}

__device__ __forceinline__ v16bf mkfrag(H8 lo, H8 hi){
  union { unsigned int u[8]; v16bf v; } t;
  t.u[0]=lo.u[0]; t.u[1]=lo.u[1]; t.u[2]=lo.u[2]; t.u[3]=lo.u[3];
  t.u[4]=hi.u[0]; t.u[5]=hi.u[1]; t.u[6]=hi.u[2]; t.u[7]=hi.u[3];
  return t.v;
}

// ---- weight conversion: pack W0[:,256:], W1..W4 as bf16 row-major into ws ----
// offsets (bf16 elems): W0b@0 (256x256), W1@65536 (128x256), W2@98304 (64x128),
//                       W3@106496 (32x64), W4@108544 (16x32). total 109056.
__global__ void prep_weights(const float* __restrict__ w0, const float* __restrict__ w1,
                             const float* __restrict__ w2, const float* __restrict__ w3,
                             const float* __restrict__ w4, __bf16* __restrict__ wb){
  int i = blockIdx.x * 256 + threadIdx.x;
  if (i < 65536){ int o = i >> 8, c = i & 255; wb[i] = (__bf16)w0[o*512 + 256 + c]; }
  else if (i < 98304)  wb[i] = (__bf16)w1[i - 65536];
  else if (i < 106496) wb[i] = (__bf16)w2[i - 98304];
  else if (i < 108544) wb[i] = (__bf16)w3[i - 106496];
  else if (i < 109056) wb[i] = (__bf16)w4[i - 108544];
}

// ---- main kernel: one block per (b,ky,kx) group (2312 candidate + 8 template) ----
__global__ void __launch_bounds__(512)
corr_group_kernel(const float* __restrict__ tmpl, const float* __restrict__ cand,
                  const float* __restrict__ w0f, const float* __restrict__ w5f,
                  const float* __restrict__ w6f, const __bf16* __restrict__ wb,
                  float* __restrict__ out)
{
  extern __shared__ char smem[];
  __bf16* Xs    = (__bf16*)smem;                         // [NPAD][CPITCH] bf16
  __bf16* Ws    = (__bf16*)(smem + WS_LDS_OFF);          // staged W1..W4 (bf16)
  float*  meanS = (float*)(smem + MISC_OFF);             // 256
  float*  biasS = meanS + 256;                           // 256
  float*  logitS= biasS + 256;                           // 240
  float*  wnS   = logitS + NPAD;                         // 256
  float*  redS  = wnS + 256;                             // 16

  const int g   = blockIdx.x;
  const int tid = threadIdx.x;
  const bool isT = (g >= 2312);
  int b, ky, kx;
  if (isT){ b = g - 2312; ky = 0; kx = 0; }
  else { b = g / 289; int r = g - b*289; ky = r / 17; kx = r - ky*17; }

  // -------- Phase A0: TDM DMA: stage W1..W4 (87KB contiguous) from ws into LDS --------
  if (tid == 0){
    unsigned long long ga = (unsigned long long)(const void*)(wb + 65536);
    u32x4 g0;
    g0[0] = 1u;                                            // count=1 valid descriptor
    g0[1] = (unsigned)WS_LDS_OFF;                          // lds_addr
    g0[2] = (unsigned)(ga & 0xffffffffull);                // global_addr[31:0]
    g0[3] = (unsigned)((ga >> 32) & 0x01ffffffull) | (2u << 30);  // addr[56:32] | type=2
    i32x8 g1;
    g1[0] = (2 << 16);                                     // data_size = 4B dwords
    g1[1] = (int)(21760u << 16);                           // tensor_dim0[15:0]=21760
    g1[2] = (1 << 16);                                     // dim0[31:16]=0, tensor_dim1=1
    g1[3] = (int)(21760u << 16);                           // tile_dim0 = 21760 dwords
    g1[4] = 1;                                             // tile_dim1 = 1
    g1[5] = 21760;                                         // tensor_dim0_stride
    g1[6] = 0; g1[7] = 0;
    i32x4 z4 = {0,0,0,0};
#if TDM_6ARG
    i32x8 z8 = {0,0,0,0,0,0,0,0};
    __builtin_amdgcn_tensor_load_to_lds(g0, g1, z4, z4, z8, 0);
#else
    __builtin_amdgcn_tensor_load_to_lds(g0, g1, z4, z4, 0);
#endif
  }

  // -------- Phase A: stage window into LDS as bf16 [pix][ch] --------
  if (isT){
    const float* base = tmpl + (size_t)b * 256 * 225;
    for (int idx = tid; idx < 256*225; idx += 512){
      int c = idx / 225, p = idx - c*225;
      Xs[p*CPITCH + c] = (__bf16)base[c*225 + p];
    }
  } else {
    const float* base = cand + (size_t)b * 256 * 961;
    for (int idx = tid; idx < 256*225; idx += 512){
      int c = idx / 225, p = idx - c*225;
      int y = p / 15, x = p - y*15;
      Xs[p*CPITCH + c] = (__bf16)base[c*961 + (y+ky)*31 + (x+kx)];
    }
  }
  for (int idx = tid; idx < (NPAD-NPIX)*256; idx += 512){   // zero pad pixels
    int p = NPIX + (idx >> 8), c = idx & 255;
    Xs[p*CPITCH + c] = (__bf16)0.f;
  }
  __syncthreads();

  // -------- Phase A2: per-channel mean over the window --------
  if (tid < 256){
    float s = 0.f;
    for (int p = 0; p < NPIX; ++p) s += (float)Xs[p*CPITCH + tid];
    meanS[tid] = s * (1.0f/225.0f);
  }
  __syncthreads();

  // -------- Phase B: layer-0 bias = W0[:, :256] @ mean (fp32) --------
  if (tid < 256){
    float s = 0.f;
    const float* wr = w0f + tid*512;
    for (int c = 0; c < 256; ++c) s += wr[c] * meanS[c];
    biasS[tid] = s;
  }
  if (tid == 0) __builtin_amdgcn_s_wait_tensorcnt(0);  // W1..W4 now resident in LDS
  __syncthreads();

  // -------- Phase C: per-wave WMMA MLP (16 pixels per wave) --------
  const int wav  = tid >> 5;
  const int lane = tid & 31;
  const int h    = lane >> 4;      // which 16-lane half
  const int pcol = lane & 15;      // pixel column within tile / weight row within M-tile
  if (wav < 15){
    const int pix = wav*16 + pcol;
    const __bf16* xrow = Xs + pix*CPITCH;
    const __bf16* w0p  = wb + pcol*256 + h*16;      // + mt*4096 + kt*32
    const __bf16* Wl1 = Ws;                          // 128x256
    const __bf16* Wl2 = Ws + 32768;                  // 64x128
    const __bf16* Wl3 = Ws + 40960;                  // 32x64
    const __bf16* Wl4 = Ws + 43008;                  // 16x32
    const v8f zero = {0,0,0,0,0,0,0,0};

    // L0: out(256x16) = W0b(256x256) @ X(256x16), K=256 (8 ksteps), 16 M-tiles.
    // Software-pipelined in chunks of 4 fragments to keep global loads in flight.
    v8f acc0[16];
    #pragma unroll
    for (int m = 0; m < 16; ++m) acc0[m] = zero;
    #pragma unroll
    for (int kt = 0; kt < 8; ++kt){
      v16bf bf = *(const v16bf*)(xrow + kt*32 + h*16);
      v16bf cur[4], nxt[4];
      #pragma unroll
      for (int q = 0; q < 4; ++q)
        cur[q] = *(const v16bf*)(w0p + kt*32 + q*4096);
      #pragma unroll
      for (int c4 = 0; c4 < 4; ++c4){
        #pragma unroll
        for (int q = 0; q < 4; ++q)
          if (c4 < 3) nxt[q] = *(const v16bf*)(w0p + kt*32 + (c4+1)*16384 + q*4096);
        #pragma unroll
        for (int q = 0; q < 4; ++q)
          acc0[c4*4+q] = __builtin_amdgcn_wmma_f32_16x16x32_bf16(false, cur[q], false, bf,
                                                                 (short)0, acc0[c4*4+q], false, false);
        #pragma unroll
        for (int q = 0; q < 4; ++q)
          if (c4 < 3) cur[q] = nxt[q];
      }
    }
    H8 hv0[16];
    #pragma unroll
    for (int mt = 0; mt < 16; ++mt){
      float v[8];
      #pragma unroll
      for (int r = 0; r < 8; ++r) v[r] = leaky(acc0[mt][r] + biasS[mt*16 + 8*h + r]);
      #pragma unroll
      for (int k = 0; k < 4; ++k) hv0[mt].u[k] = pk2(v[2*k], v[2*k+1]);
    }

    // L1: 256 -> 128 (weights from LDS)
    v8f acc1[8];
    #pragma unroll
    for (int m = 0; m < 8; ++m) acc1[m] = zero;
    #pragma unroll
    for (int kt = 0; kt < 8; ++kt){
      H8 own = hv0[2*kt + h], oth = shfl16(hv0[2*kt + 1 - h]);
      v16bf bf = h ? mkfrag(oth, own) : mkfrag(own, oth);
      #pragma unroll
      for (int mt = 0; mt < 8; ++mt){
        v16bf af = *(const v16bf*)(Wl1 + (mt*16 + pcol)*256 + kt*32 + h*16);
        acc1[mt] = __builtin_amdgcn_wmma_f32_16x16x32_bf16(false, af, false, bf,
                                                           (short)0, acc1[mt], false, false);
      }
    }
    H8 hv1[8];
    #pragma unroll
    for (int mt = 0; mt < 8; ++mt){
      float v[8];
      #pragma unroll
      for (int r = 0; r < 8; ++r) v[r] = leaky(acc1[mt][r]);
      #pragma unroll
      for (int k = 0; k < 4; ++k) hv1[mt].u[k] = pk2(v[2*k], v[2*k+1]);
    }

    // L2: 128 -> 64
    v8f acc2[4];
    #pragma unroll
    for (int m = 0; m < 4; ++m) acc2[m] = zero;
    #pragma unroll
    for (int kt = 0; kt < 4; ++kt){
      H8 own = hv1[2*kt + h], oth = shfl16(hv1[2*kt + 1 - h]);
      v16bf bf = h ? mkfrag(oth, own) : mkfrag(own, oth);
      #pragma unroll
      for (int mt = 0; mt < 4; ++mt){
        v16bf af = *(const v16bf*)(Wl2 + (mt*16 + pcol)*128 + kt*32 + h*16);
        acc2[mt] = __builtin_amdgcn_wmma_f32_16x16x32_bf16(false, af, false, bf,
                                                           (short)0, acc2[mt], false, false);
      }
    }
    H8 hv2[4];
    #pragma unroll
    for (int mt = 0; mt < 4; ++mt){
      float v[8];
      #pragma unroll
      for (int r = 0; r < 8; ++r) v[r] = leaky(acc2[mt][r]);
      #pragma unroll
      for (int k = 0; k < 4; ++k) hv2[mt].u[k] = pk2(v[2*k], v[2*k+1]);
    }

    // L3: 64 -> 32
    v8f acc3[2];
    acc3[0] = zero; acc3[1] = zero;
    #pragma unroll
    for (int kt = 0; kt < 2; ++kt){
      H8 own = hv2[2*kt + h], oth = shfl16(hv2[2*kt + 1 - h]);
      v16bf bf = h ? mkfrag(oth, own) : mkfrag(own, oth);
      #pragma unroll
      for (int mt = 0; mt < 2; ++mt){
        v16bf af = *(const v16bf*)(Wl3 + (mt*16 + pcol)*64 + kt*32 + h*16);
        acc3[mt] = __builtin_amdgcn_wmma_f32_16x16x32_bf16(false, af, false, bf,
                                                           (short)0, acc3[mt], false, false);
      }
    }
    H8 hv3[2];
    #pragma unroll
    for (int mt = 0; mt < 2; ++mt){
      float v[8];
      #pragma unroll
      for (int r = 0; r < 8; ++r) v[r] = leaky(acc3[mt][r]);
      #pragma unroll
      for (int k = 0; k < 4; ++k) hv3[mt].u[k] = pk2(v[2*k], v[2*k+1]);
    }

    // L4: 32 -> 16 (single WMMA)
    v8f acc4 = zero;
    {
      H8 own = hv3[h], oth = shfl16(hv3[1 - h]);
      v16bf bf = h ? mkfrag(oth, own) : mkfrag(own, oth);
      v16bf af = *(const v16bf*)(Wl4 + pcol*32 + h*16);
      acc4 = __builtin_amdgcn_wmma_f32_16x16x32_bf16(false, af, false, bf,
                                                     (short)0, acc4, false, false);
    }
    float a4[8];
    #pragma unroll
    for (int r = 0; r < 8; ++r) a4[r] = leaky(acc4[r]);   // channel 8h+r of pixel pcol

    // L5: 16 -> 8 (fp32, half-channel partials + cross-half shuffle)
    float p5[8];
    #pragma unroll
    for (int o = 0; o < 8; ++o){
      float s = 0.f;
      #pragma unroll
      for (int r = 0; r < 8; ++r) s += w5f[o*16 + 8*h + r] * a4[r];
      p5[o] = s;
    }
    #pragma unroll
    for (int o = 0; o < 8; ++o){
      p5[o] += __shfl_xor(p5[o], 16, 32);
      p5[o]  = leaky(p5[o]);
    }
    // L6: 8 -> 1 (no activation)
    float lg = 0.f;
    #pragma unroll
    for (int c = 0; c < 8; ++c) lg += w6f[c] * p5[c];
    if (h == 0) logitS[pix] = lg;
  }
  __syncthreads();

  // -------- Phase D: softmax over 225 pixels (wave-parallel reductions) --------
  {
    float v = -3.0e38f;
    if (tid < NPIX) v = logitS[tid];
    if (tid < 256){
      #pragma unroll
      for (int off = 16; off; off >>= 1) v = fmaxf(v, __shfl_xor(v, off, 32));
      if ((tid & 31) == 0) redS[tid >> 5] = v;
    }
    __syncthreads();
    if (tid == 0){
      float m = redS[0];
      #pragma unroll
      for (int i = 1; i < 8; ++i) m = fmaxf(m, redS[i]);
      redS[8] = m;
    }
    __syncthreads();
    float e = 0.f;
    if (tid < NPIX) e = __expf(logitS[tid] - redS[8]);
    if (tid < 256){
      wnS[tid] = e;
      float s = e;
      #pragma unroll
      for (int off = 16; off; off >>= 1) s += __shfl_xor(s, off, 32);
      if ((tid & 31) == 0) redS[tid >> 5] = s;
    }
    __syncthreads();
    if (tid == 0){
      float s = 0.f;
      #pragma unroll
      for (int i = 0; i < 8; ++i) s += redS[i];
      redS[8] = 1.0f / s;
    }
    __syncthreads();
  }

  // -------- Phase E: weighted spatial sum per channel --------
  if (tid < 256){
    float inv = redS[8];
    float s = 0.f;
    for (int p = 0; p < NPIX; ++p) s += (float)Xs[p*CPITCH + tid] * (wnS[p] * inv);
    if (isT) out[b*256 + tid] = s;
    else     out[2048 + (size_t)(b*256 + tid)*289 + ky*17 + kx] = s;
  }
}

extern "C" void kernel_launch(void* const* d_in, const int* in_sizes, int n_in,
                              void* d_out, int out_size, void* d_ws, size_t ws_size,
                              hipStream_t stream)
{
  (void)in_sizes; (void)n_in; (void)out_size; (void)ws_size;
  const float* tmpl = (const float*)d_in[0];
  const float* cand = (const float*)d_in[1];
  const float* w0 = (const float*)d_in[2];
  const float* w1 = (const float*)d_in[3];
  const float* w2 = (const float*)d_in[4];
  const float* w3 = (const float*)d_in[5];
  const float* w4 = (const float*)d_in[6];
  const float* w5 = (const float*)d_in[7];
  const float* w6 = (const float*)d_in[8];
  __bf16* wb = (__bf16*)d_ws;

  prep_weights<<<(109056 + 255) / 256, 256, 0, stream>>>(w0, w1, w2, w3, w4, wb);

  size_t smem = (size_t)XS_BYTES + WS_BYTES + 1040 * sizeof(float);
  corr_group_kernel<<<2320, 512, smem, stream>>>(tmpl, cand, w0, w5, w6, wb, (float*)d_out);
}
